// CausalSelfAttention_63436666962248
// MI455X (gfx1250) — matrix operand order repaired and checked by reading, not compiled, not run on hardware
//
#include <hip/hip_runtime.h>
#include <hip/hip_bf16.h>
#include <stdint.h>

// Problem dims (fixed by the reference)
#define BB  4
#define NN  2048
#define DDIM 1024
#define HH  16
#define DKK 64
#define DVV 64

// Wave macro-tile: 32 rows x 64 cols of C  (TM x TN subtiles of 16x16)
#define TM 2
#define TN 4

typedef __attribute__((ext_vector_type(8)))  float  v8f;
typedef __attribute__((ext_vector_type(16))) __bf16 v16bf;

union FragBF {
  uint4 u[2];   // 8 VGPRs worth: dwords 0..3 and 4..7
  v16bf v;      // 16 bf16 elements
};

static __device__ __forceinline__ unsigned short f2bf(float f) {
  union { float f; unsigned int u; } x; x.f = f;
  unsigned int u = x.u;
  unsigned int r = (u + 0x7FFFu + ((u >> 16) & 1u)) >> 16;  // RNE
  return (unsigned short)r;
}

// ---------------- converts ----------------

__global__ void k_f32_to_bf16(const float* __restrict__ in,
                              unsigned short* __restrict__ out, long n) {
  long stride = (long)gridDim.x * blockDim.x;
  for (long t = (long)blockIdx.x * blockDim.x + threadIdx.x; t < n; t += stride)
    out[t] = f2bf(in[t]);
}

// W: (Hh, Dd, Kk) fp32 row-major  ->  WT: (Hh, Kk, Dd) bf16 (K-major operand)
__global__ void k_f32_to_bf16_T(const float* __restrict__ W,
                                unsigned short* __restrict__ WT,
                                int Hh, int Dd, int Kk) {
  long n = (long)Hh * Dd * Kk;
  long stride = (long)gridDim.x * blockDim.x;
  for (long t = (long)blockIdx.x * blockDim.x + threadIdx.x; t < n; t += stride) {
    int dk = Dd * Kk;
    int h  = (int)(t / dk);
    int r  = (int)(t % dk);
    int d  = r / Kk;
    int k  = r - d * Kk;
    WT[(long)h * dk + (long)k * Dd + d] = f2bf(W[t]);
  }
}

// ---------------- generic bf16 WMMA GEMM, 32x64 macro-tile per wave ----------
// C(M x Nn) = A(M x K, row-major bf16, ld=LDA) * B(K x Nn), B supplied
// K-contiguous as Bt(Nn x K, row-major bf16, ld=LDB).
// LDA/LDB/LDC/MODE are template constants so all subtile addresses fold into
// base + 24-bit immediate offsets (2 base pointer pairs total, no spills).
// MODE: 0 = fp32 row-major store, 1 = bf16 row-major, 2 = bf16 transposed.
// Requires: M % 32 == 0, Nn % 64 == 0, K % 32 == 0.
template <int LDA, int LDB, int LDC, int MODE>
__global__ void k_gemm_bf16_wmma(const unsigned short* __restrict__ A,
                                 long As0, long As1,
                                 const unsigned short* __restrict__ Bt,
                                 long Bs0, long Bs1,
                                 void* __restrict__ C,
                                 long Cs0, long Cs1,
                                 int M, int Nn, int K, int div) {
  int e  = blockIdx.y;
  int i0 = e / div;
  int i1 = e - i0 * div;
  const unsigned short* Ab = A  + i0 * As0 + i1 * As1;
  const unsigned short* Bb = Bt + i0 * Bs0 + i1 * Bs1;

  int lane = (int)(threadIdx.x & 31u);
  int wave = (int)(threadIdx.x >> 5);
  int tiles_n = Nn >> 6;                       // 64-wide macro-columns
  int tile = blockIdx.x * 4 + wave;            // 4 waves / 128-thread block
  if (tile >= (M >> 5) * tiles_n) return;      // wave-uniform exit
  int mt   = tile / tiles_n;
  int mrow = mt << 5;
  int ncol = (tile - mt * tiles_n) << 6;

  int lm   = lane & 15;   // A: M row ; B/C: N column
  int half = lane >> 4;

  // Single base pointer per operand; subtiles via constant immediate offsets.
  const unsigned short* abase = Ab + (long)(mrow + lm) * LDA;
  const unsigned short* bbase = Bb + (long)(ncol + lm) * LDB;

  v8f acc[TM][TN];
  #pragma unroll
  for (int i = 0; i < TM; ++i)
    #pragma unroll
    for (int j = 0; j < TN; ++j) { v8f z = {}; acc[i][j] = z; }

  for (int kb = 0; kb < K; kb += 32) {
    FragBF a[TM], b[TN];
    // A 16x32 bf16 layout: VGPR0-3 <- K[8*half .. +7], VGPR4-7 <- K[16+8*half .. +7]
    #pragma unroll
    for (int i = 0; i < TM; ++i) {
      a[i].u[0] = *(const uint4*)(abase + 16 * i * LDA + kb + 8 * half);
      a[i].u[1] = *(const uint4*)(abase + 16 * i * LDA + kb + 16 + 8 * half);
    }
    // B 32x16 bf16 layout (lane=N col): VGPR0-3 <- K[8*half .. +7], VGPR4-7 <- K[16+8*half .. +7]
    #pragma unroll
    for (int j = 0; j < TN; ++j) {
      b[j].u[0] = *(const uint4*)(bbase + 16 * j * LDB + kb + 8 * half);
      b[j].u[1] = *(const uint4*)(bbase + 16 * j * LDB + kb + 16 + 8 * half);
    }
    __builtin_prefetch(abase + kb + 128, 0, 0);   // global_prefetch_b8
    __builtin_prefetch(bbase + kb + 128, 0, 0);
    #pragma unroll
    for (int i = 0; i < TM; ++i)
      #pragma unroll
      for (int j = 0; j < TN; ++j)
        acc[i][j] = __builtin_amdgcn_wmma_f32_16x16x32_bf16(
            /*neg_a=*/false, a[i].v, /*neg_b=*/false, b[j].v,
            /*c_mod=*/(short)0, acc[i][j], /*reuse_a=*/false, /*reuse_b=*/false);
  }

  // C/D layout: lane L -> N = L&15 ; VGPR r -> M = r + 8*(L>>4)
  if (MODE == 0) {
    float* Cf = (float*)C + i0 * Cs0 + i1 * Cs1 + (long)(mrow + 8 * half) * LDC
                + ncol + lm;
    #pragma unroll
    for (int i = 0; i < TM; ++i)
      #pragma unroll
      for (int j = 0; j < TN; ++j)
        #pragma unroll
        for (int r = 0; r < 8; ++r)
          Cf[(16 * i + r) * LDC + 16 * j] = acc[i][j][r];
  } else if (MODE == 1) {
    unsigned short* Ch = (unsigned short*)C + i0 * Cs0 + i1 * Cs1
                         + (long)(mrow + 8 * half) * LDC + ncol + lm;
    #pragma unroll
    for (int i = 0; i < TM; ++i)
      #pragma unroll
      for (int j = 0; j < TN; ++j)
        #pragma unroll
        for (int r = 0; r < 8; ++r)
          Ch[(16 * i + r) * LDC + 16 * j] = f2bf(acc[i][j][r]);
  } else {
    unsigned short* Ch = (unsigned short*)C + i0 * Cs0 + i1 * Cs1
                         + (long)(ncol + lm) * LDC + mrow + 8 * half;
    #pragma unroll
    for (int i = 0; i < TM; ++i)
      #pragma unroll
      for (int j = 0; j < TN; ++j)
        #pragma unroll
        for (int r = 0; r < 8; ++r)   // transposed: contiguous along M per lane
          Ch[16 * j * LDC + 16 * i + r] = f2bf(acc[i][j][r]);
  }
}

// ---------------- column softmax (over the QUERY axis, per reference quirk) --

__global__ void k_col_softmax_stats(const float* __restrict__ S,
                                    float* __restrict__ cmax,
                                    float* __restrict__ crcp, int n) {
  int j = blockIdx.x * blockDim.x + threadIdx.x;   // column (key index)
  if (j >= n) return;
  float mx = -3.402823466e38f;
  for (int i = 0; i < n; ++i) mx = fmaxf(mx, S[(long)i * n + j]);
  float s = 0.0f;
  for (int i = 0; i < n; ++i) s += __expf(S[(long)i * n + j] - mx);
  cmax[j] = mx;
  crcp[j] = 1.0f / s;
}

__global__ void k_softmax_norm_bf16(const float* __restrict__ S,
                                    const float* __restrict__ cmax,
                                    const float* __restrict__ crcp,
                                    unsigned short* __restrict__ Aout, int n) {
  long total  = (long)n * n;
  long stride = (long)gridDim.x * blockDim.x;
  for (long t = (long)blockIdx.x * blockDim.x + threadIdx.x; t < total; t += stride) {
    int j = (int)(t % n);
    Aout[t] = f2bf(__expf(S[t] - cmax[j]) * crcp[j]);
  }
}

// ---------------- host orchestration ----------------

extern "C" void kernel_launch(void* const* d_in, const int* in_sizes, int n_in,
                              void* d_out, int out_size, void* d_ws, size_t ws_size,
                              hipStream_t stream) {
  (void)in_sizes; (void)n_in; (void)out_size; (void)ws_size;
  const float* X  = (const float*)d_in[0];
  const float* Wq = (const float*)d_in[1];
  const float* Wk = (const float*)d_in[2];
  const float* Wv = (const float*)d_in[3];
  const float* Wo = (const float*)d_in[4];
  float* Out = (float*)d_out;

  char* base = (char*)d_ws;
  auto alloc = [&](size_t bytes) -> char* {
    char* p = base;
    base += (bytes + 255) & ~(size_t)255;
    return p;
  };

  const long nX  = (long)BB * NN * DDIM;        // 8,388,608
  const long nW  = (long)HH * DDIM * DKK;       // 1,048,576
  const long nWo = (long)DDIM * DDIM;           // 1,048,576
  const long nQ  = (long)BB * HH * NN * DKK;    // 8,388,608
  const long nS  = (long)NN * NN;               // 4,194,304

  unsigned short* Xh  = (unsigned short*)alloc(nX  * 2);  // X in bf16
  unsigned short* WqT = (unsigned short*)alloc(nW  * 2);  // (H, dk, D)
  unsigned short* WkT = (unsigned short*)alloc(nW  * 2);
  unsigned short* WvT = (unsigned short*)alloc(nW  * 2);
  unsigned short* WoT = (unsigned short*)alloc(nWo * 2);  // K-major W_O
  unsigned short* Qb  = (unsigned short*)alloc(nQ  * 2);  // (B,H,N,dk)
  unsigned short* Kb  = (unsigned short*)alloc(nQ  * 2);  // (B,H,N,dk) == Kt for S
  unsigned short* Vt  = (unsigned short*)alloc(nQ  * 2);  // (B,H,dv,N) transposed
  unsigned short* AVb = (unsigned short*)alloc(nQ  * 2);  // (B,H,N,dv) contiguous
  unsigned short* Abf = (unsigned short*)alloc(nS  * 2);  // one (b,h) A in bf16
  float* Sf   = (float*)alloc(nS * 4);                    // one (b,h) score matrix
  float* cmax = (float*)alloc(NN * 4);
  float* crcp = (float*)alloc(NN * 4);

  // -- precision converts (weights stored K-contiguous for the WMMA B operand)
  k_f32_to_bf16 <<<4096, 256, 0, stream>>>(X, Xh, nX);
  k_f32_to_bf16_T<<<2048, 256, 0, stream>>>(Wq, WqT, HH, DDIM, DKK);
  k_f32_to_bf16_T<<<2048, 256, 0, stream>>>(Wk, WkT, HH, DDIM, DKK);
  k_f32_to_bf16_T<<<2048, 256, 0, stream>>>(Wv, WvT, HH, DDIM, DKK);
  k_f32_to_bf16_T<<<2048, 256, 0, stream>>>(Wo, WoT, 1, DDIM, DDIM);

  // -- batched projections: (2048/32)*(64/64)=64 waves per (b,h) -> 16 blocks
  {
    dim3 grid(16, BB * HH);
    k_gemm_bf16_wmma<DDIM, DDIM, DKK, 1><<<grid, 128, 0, stream>>>(   // Q
        Xh, (long)NN * DDIM, 0,
        WqT, 0, (long)DKK * DDIM,
        Qb, (long)HH * NN * DKK, (long)NN * DKK,
        NN, DKK, DDIM, HH);
    k_gemm_bf16_wmma<DDIM, DDIM, DKK, 1><<<grid, 128, 0, stream>>>(   // K
        Xh, (long)NN * DDIM, 0,
        WkT, 0, (long)DKK * DDIM,
        Kb, (long)HH * NN * DKK, (long)NN * DKK,
        NN, DKK, DDIM, HH);
    k_gemm_bf16_wmma<DDIM, DDIM, NN, 2><<<grid, 128, 0, stream>>>(    // V -> Vt
        Xh, (long)NN * DDIM, 0,
        WvT, 0, (long)DVV * DDIM,
        Vt, (long)HH * DVV * NN, (long)DVV * NN,
        NN, DVV, DDIM, HH);
  }

  // -- per-(b,h) attention: S -> column softmax -> AV
  for (int e = 0; e < BB * HH; ++e) {
    const unsigned short* Qe = Qb + (long)e * NN * DKK;
    const unsigned short* Ke = Kb + (long)e * NN * DKK;   // already Kᵀ K-major
    const unsigned short* Ve = Vt + (long)e * DVV * NN;
    unsigned short* AVe = AVb + (long)e * NN * DVV;

    // S = Q Kᵀ (fp32): (2048/32)*(2048/64) = 2048 waves -> 512 blocks
    k_gemm_bf16_wmma<DKK, DKK, NN, 0><<<dim3(512, 1), 128, 0, stream>>>(
        Qe, 0, 0,
        Ke, 0, 0,
        Sf, 0, 0,
        NN, NN, DKK, 1);
    k_col_softmax_stats<<<NN / 256, 256, 0, stream>>>(Sf, cmax, crcp, NN);
    k_softmax_norm_bf16<<<4096, 256, 0, stream>>>(Sf, cmax, crcp, Abf, NN);
    // AV = A V: 64 waves -> 16 blocks
    k_gemm_bf16_wmma<NN, NN, DVV, 1><<<dim3(16, 1), 128, 0, stream>>>(
        Abf, 0, 0,
        Ve, 0, 0,
        AVe, 0, 0,
        NN, DVV, NN, 1);
  }

  // -- output projection: raw-reshape quirk == view AVb[b] as (N x H*dv)
  //    (2048/32)*(1024/64) = 1024 waves -> 256 blocks, batched over B
  k_gemm_bf16_wmma<DDIM, DDIM, DDIM, 0><<<dim3(256, BB), 128, 0, stream>>>(
      AVb, (long)NN * HH * DVV, 0,
      WoT, 0, 0,
      Out, (long)NN * DDIM, 0,
      NN, DDIM, HH * DVV, 1);
}
